// UnetGraphSAGE_67405216743478
// MI455X (gfx1250) — matrix operand order repaired and checked by reading, not compiled, not run on hardware
//
#include <hip/hip_runtime.h>
#include <hip/hip_bf16.h>

typedef __attribute__((ext_vector_type(16))) _Float16 v16h;
typedef __attribute__((ext_vector_type(8)))  float    v8f;
typedef __attribute__((ext_vector_type(4)))  float    v4f;

// ---------------------------------------------------------------------------
// WMMA GEMM:  Y[N, Ktot] = A[N, cin] (f32) x W[cin, Ktot] (f16, stored as
// WT[Ktot][cin] transposed).  One 16x16 tile per wave, 4 waves per block.
// The A tile (16 rows x 32 k, f32) is shared by all 4 waves, so it is staged
// once per block in LDS via CDNA5 async-to-LDS copies (ASYNCcnt-tracked,
// no VGPR round trip), then each lane builds its WMMA A-fragment from LDS
// and converts f32->f16.  k-step = 32 with v_wmma_f32_16x16x32_f16.
// Grid exactly tiles the output (N%16==0, Ktot%64==0, cin%32==0 for every
// layer) so EXEC is all-ones at every WMMA as the ISA requires.
// ---------------------------------------------------------------------------
__global__ __launch_bounds__(128)
void wmma_gemm_kernel(const float* __restrict__ A,
                      const _Float16* __restrict__ WT,
                      float* __restrict__ Y,
                      int cin, int Ktot) {
  __shared__ float tileA[16 * 32];   // 2 KB: one k-step of the 16-row A tile

  const int t = threadIdx.x;
  const int lane = t & 31;
  const int wave = t >> 5;
  const int m = lane & 15;        // row (A) / col (B) within tile
  const int h = lane >> 4;        // lane-half selects K sub-block
  const long row0 = (long)blockIdx.x * 16;
  const long col0 = (long)blockIdx.y * 64 + (long)wave * 16;

  // cooperative async stage: 128 threads x 16B = 2048B = whole tile
  const int arow = t >> 3;        // 0..15
  const int achk = t & 7;         // 16B chunk within the 128B row
  const float* agsrc = A + (row0 + arow) * (long)cin + achk * 4;
  // addrspace(3) offset lives in the low 32 bits of the flat address
  const unsigned lds_dst =
      (unsigned)(unsigned long long)(const void*)&tileA[0] + (unsigned)t * 16u;

  const _Float16* brow = WT + (col0 + m) * (long)cin;
  // A fragment source in LDS: elem i -> K = i + 16*h
  const float* afrag0 = &tileA[m * 32 + 16 * h];
  const float* afrag1 = &tileA[m * 32 + 8 + 16 * h];

  v8f acc = {};
  for (int k0 = 0; k0 < cin; k0 += 32) {
    __syncthreads();              // previous tile fully consumed
    asm volatile("global_load_async_to_lds_b128 %0, %1, off"
                 :: "v"(lds_dst), "v"(agsrc + k0) : "memory");
    asm volatile("s_wait_asynccnt 0x0" ::: "memory");
    __syncthreads();              // whole tile visible to all 4 waves

    v4f a0 = *(const v4f*)(afrag0);
    v4f a1 = *(const v4f*)(afrag0 + 4);
    v4f a2 = *(const v4f*)(afrag1);
    v4f a3 = *(const v4f*)(afrag1 + 4);
    v16h af;
#pragma unroll
    for (int i = 0; i < 4; ++i) {
      af[i]      = (_Float16)a0[i];
      af[4 + i]  = (_Float16)a1[i];
      af[8 + i]  = (_Float16)a2[i];
      af[12 + i] = (_Float16)a3[i];
    }
    // B fragment: elem i -> B[k0 + i + 16*h][col] = WT[col][k0 + i + 16*h]
    v16h bf = *(const v16h*)(brow + k0 + 16 * h);

    acc = __builtin_amdgcn_wmma_f32_16x16x32_f16(
        /*neg_a=*/false, af, /*neg_b=*/false, bf,
        /*c_mod=*/(short)0, acc, /*reuse_a=*/false, /*reuse_b=*/false);
  }
  // C/D layout: VGPR r, lanes 0-15 -> M = r, lanes 16-31 -> M = r + 8
#pragma unroll
  for (int r = 0; r < 8; ++r)
    Y[(row0 + r + 8 * h) * (long)Ktot + col0 + m] = acc[r];
}

// Pack conv weights: WT[k][i] f16, k in [0,3*cout):
//   k <  2*cout : basis[b][i][o]  (b = k/cout, o = k%cout)
//   k >= 2*cout : w_self[i][o]    (o = k - 2*cout)
__global__ void pack_conv_w(const float* __restrict__ basis,
                            const float* __restrict__ wself,
                            _Float16* __restrict__ WT,
                            int cin, int cout, long tot) {
  long idx = (long)blockIdx.x * 256 + threadIdx.x;
  if (idx >= tot) return;
  int i = (int)(idx % cin);
  int k = (int)(idx / cin);
  float v;
  if (k < 2 * cout) {
    int b = k / cout, o = k % cout;
    v = basis[((long)b * cin + i) * cout + o];
  } else {
    int o = k - 2 * cout;
    v = wself[(long)i * cout + o];
  }
  WT[idx] = (_Float16)v;
}

// Pack ConvTranspose2d weights w[c][o][a][b] -> WT[ab*C + o][c]
__global__ void pack_up_w(const float* __restrict__ w,
                          _Float16* __restrict__ WT, int C, long tot) {
  long idx = (long)blockIdx.x * 256 + threadIdx.x;
  if (idx >= tot) return;
  int c = (int)(idx % C);
  long k = idx / C;
  int o = (int)(k % C);
  int ab = (int)(k / C);
  WT[idx] = (_Float16)w[((long)c * C + o) * 4 + ab];
}

// out[n][c] = Y[n][2*cout + c] + bias[c]   (self-loop + bias, pre-aggregation)
__global__ void self_bias_init(const float* __restrict__ Y,
                               const float* __restrict__ bias,
                               float* __restrict__ out,
                               int cout, int Ktot, long tot) {
  long idx = (long)blockIdx.x * 256 + threadIdx.x;
  if (idx >= tot) return;
  long n = idx / cout;
  int c = (int)(idx % cout);
  out[idx] = Y[n * (long)Ktot + 2 * cout + c] + bias[c];
}

// out[dst[e]][c] += comp[et[e]][0]*Y[src[e]][c] + comp[et[e]][1]*Y[src[e]][cout+c]
__global__ void edge_scatter(const float* __restrict__ Y,
                             const int* __restrict__ src,
                             const int* __restrict__ dst,
                             const int* __restrict__ et,
                             const float* __restrict__ comp,
                             float* __restrict__ out,
                             int cout, int Ktot, long tot) {
  long idx = (long)blockIdx.x * 256 + threadIdx.x;
  if (idx >= tot) return;
  long e = idx / cout;
  int c = (int)(idx % cout);
  int s = src[e], d = dst[e], t = et[e];
  float c0 = comp[t * 2], c1 = comp[t * 2 + 1];
  float v = c0 * Y[(long)s * Ktot + c] + c1 * Y[(long)s * Ktot + cout + c];
  atomicAdd(out + (long)d * cout + c, v);
}

__global__ void relu_k(float* __restrict__ x, long tot) {
  long idx = (long)blockIdx.x * 256 + threadIdx.x;
  if (idx >= tot) return;
  x[idx] = fmaxf(x[idx], 0.0f);
}

// 2x2 average pool per face; node = face*res*res + y*res + x, channel-last
__global__ void avg_pool_k(const float* __restrict__ in, float* __restrict__ out,
                           int res, int C, long tot) {
  long idx = (long)blockIdx.x * 256 + threadIdx.x;
  if (idx >= tot) return;
  int r2 = res >> 1;
  int c = (int)(idx % C);
  long pn = idx / C;
  int face = (int)(pn / ((long)r2 * r2));
  int rem = (int)(pn % ((long)r2 * r2));
  int y = rem / r2, x = rem % r2;
  long b = ((long)face * res + 2 * y) * res + 2 * x;
  float v = in[b * C + c] + in[(b + 1) * C + c] +
            in[(b + res) * C + c] + in[(b + res + 1) * C + c];
  out[pn * (long)C + c] = 0.25f * v;
}

// Place Yup[n][ab*C + o] + bias[o] at output node (face, 2i+a, 2j+b), col o
// of the concat buffer (Ctot = 2*C, up-result occupies cols [0, C)).
__global__ void up_scatter(const float* __restrict__ Yup,
                           const float* __restrict__ bias,
                           float* __restrict__ cat,
                           int res, int C, int Ctot, long tot) {
  long idx = (long)blockIdx.x * 256 + threadIdx.x;
  if (idx >= tot) return;
  int o = (int)(idx % C);
  long q = idx / C;
  int ab = (int)(q % 4);
  long n = q / 4;
  int a = ab >> 1, b = ab & 1;
  int face = (int)(n / ((long)res * res));
  int rem = (int)(n % ((long)res * res));
  int i = rem / res, j = rem % res;
  int res2 = res * 2;
  long on = ((long)face * res2 + (2 * i + a)) * res2 + (2 * j + b);
  cat[on * (long)Ctot + o] = Yup[n * (4l * C) + (long)ab * C + o] + bias[o];
}

// dst[n][colOff + c] = src[n][c]   (skip-connection half of concat)
__global__ void copy_cols(const float* __restrict__ src, float* __restrict__ dst,
                          int C, int Ctot, int colOff, long tot) {
  long idx = (long)blockIdx.x * 256 + threadIdx.x;
  if (idx >= tot) return;
  long n = idx / C;
  int c = (int)(idx % C);
  dst[n * (long)Ctot + colOff + c] = src[idx];
}

// ---------------------------------------------------------------------------
extern "C" void kernel_launch(void* const* d_in, const int* in_sizes, int n_in,
                              void* d_out, int out_size, void* d_ws, size_t ws_size,
                              hipStream_t stream) {
  (void)in_sizes; (void)n_in; (void)out_size; (void)ws_size;

  // layer table: {cin, cout} in LAYERS dict order
  static const int LC[23][2] = {
      {64, 64},   {64, 64},   {64, 128},  {128, 128}, {128, 256}, {256, 256},
      {256, 512}, {512, 512}, {512, 1024},{1024,1024},{1024,512}, {1024,512},
      {512, 512}, {512, 256}, {512, 256}, {256, 256}, {256, 128}, {256, 128},
      {128, 128}, {128, 64},  {128, 64},  {64, 64},   {64, 64}};
  const int N1 = 98304, N2 = 24576, N3 = 6144, N4 = 1536, N5 = 384;

  // ---- workspace layout -------------------------------------------------
  size_t off = 0;
  auto wsalloc = [&](size_t bytes) -> void* {
    off = (off + 255) & ~(size_t)255;
    void* p = (char*)d_ws + off;
    off += bytes;
    return p;
  };
  float*    Ybuf  = (float*)   wsalloc((size_t)N1 * 192 * sizeof(float));   // max N*Ktot
  _Float16* wtbuf = (_Float16*)wsalloc((size_t)3072 * 1024 * sizeof(_Float16));
  float*    bufA  = (float*)   wsalloc((size_t)N1 * 128 * sizeof(float));
  float*    bufB  = (float*)   wsalloc((size_t)N1 * 128 * sizeof(float));
  float*    skip1 = (float*)   wsalloc((size_t)N1 * 64  * sizeof(float));
  float*    skip2 = (float*)   wsalloc((size_t)N2 * 128 * sizeof(float));
  float*    skip3 = (float*)   wsalloc((size_t)N3 * 256 * sizeof(float));
  float*    skip4 = (float*)   wsalloc((size_t)N4 * 512 * sizeof(float));

  auto blocks = [](long tot) { return (unsigned)((tot + 255) / 256); };

  // ---- one fused RelGraphConv ------------------------------------------
  auto run_conv = [&](const float* hin, float* hout, int N, int li, int lvl,
                      bool relu_on) {
    const int cin = LC[li][0], cout = LC[li][1], Ktot = 3 * cout;
    const float* basis = (const float*)d_in[16 + 4 * li];
    const float* comp  = (const float*)d_in[16 + 4 * li + 1];
    const float* wself = (const float*)d_in[16 + 4 * li + 2];
    const float* bias  = (const float*)d_in[16 + 4 * li + 3];
    const int* src = (const int*)d_in[1 + 3 * (lvl - 1)];
    const int* dst = (const int*)d_in[2 + 3 * (lvl - 1)];
    const int* et  = (const int*)d_in[3 + 3 * (lvl - 1)];
    const long E = 4l * N;

    long tot = (long)Ktot * cin;
    pack_conv_w<<<blocks(tot), 256, 0, stream>>>(basis, wself, wtbuf, cin, cout, tot);

    dim3 g(N / 16, Ktot / 64);
    wmma_gemm_kernel<<<g, 128, 0, stream>>>(hin, wtbuf, Ybuf, cin, Ktot);

    tot = (long)N * cout;
    self_bias_init<<<blocks(tot), 256, 0, stream>>>(Ybuf, bias, hout, cout, Ktot, tot);

    tot = E * cout;
    edge_scatter<<<blocks(tot), 256, 0, stream>>>(Ybuf, src, dst, et, comp, hout,
                                                  cout, Ktot, tot);
    if (relu_on) {
      tot = (long)N * cout;
      relu_k<<<blocks(tot), 256, 0, stream>>>(hout, tot);
    }
  };

  // ---- ConvTranspose2d(k=2,s=2) + concat with skip ---------------------
  auto run_up = [&](const float* hin, const float* skip, float* cat, int j,
                    int C, int res_in) {
    const float* w    = (const float*)d_in[108 + 2 * j];
    const float* bias = (const float*)d_in[109 + 2 * j];
    const long Nin = 6l * res_in * res_in;
    const int Ktot = 4 * C, Ctot = 2 * C;

    long tot = (long)Ktot * C;
    pack_up_w<<<blocks(tot), 256, 0, stream>>>(w, wtbuf, C, tot);

    dim3 g((unsigned)(Nin / 16), Ktot / 64);
    wmma_gemm_kernel<<<g, 128, 0, stream>>>(hin, wtbuf, Ybuf, C, Ktot);

    tot = Nin * Ktot;
    up_scatter<<<blocks(tot), 256, 0, stream>>>(Ybuf, bias, cat, res_in, C, Ctot, tot);

    tot = Nin * 4 * (long)C;   // Nout * C
    copy_cols<<<blocks(tot), 256, 0, stream>>>(skip, cat, C, Ctot, C, tot);
  };

  auto run_pool = [&](const float* in, float* out, int res, int C) {
    long tot = 6l * (res / 2) * (res / 2) * C;
    avg_pool_k<<<blocks(tot), 256, 0, stream>>>(in, out, res, C, tot);
  };

  const float* in_feat = (const float*)d_in[0];
  float* out = (float*)d_out;

  // ---------------- encoder ----------------
  run_conv(in_feat, bufA, N1, 0, 1, true);    // h1
  run_conv(bufA, skip1, N1, 1, 1, true);      // h22 (skip)
  run_pool(skip1, bufB, 128, 64);             // h2
  run_conv(bufB, bufA, N2, 2, 2, true);       // h3
  run_conv(bufA, skip2, N2, 3, 2, true);      // h33 (skip)
  run_pool(skip2, bufB, 64, 128);             // h3p
  run_conv(bufB, bufA, N3, 4, 3, true);       // h4
  run_conv(bufA, skip3, N3, 5, 3, true);      // h44 (skip)
  run_pool(skip3, bufB, 32, 256);             // h4p
  run_conv(bufB, bufA, N4, 6, 4, true);       // h5
  run_conv(bufA, skip4, N4, 7, 4, true);      // h55 (skip)
  run_pool(skip4, bufB, 16, 512);             // h5p
  // ---------------- bottleneck ----------------
  run_conv(bufB, bufA, N5, 8, 5, true);       // conv6
  run_conv(bufA, bufB, N5, 9, 5, true);       // conv66
  run_conv(bufB, bufA, N5, 10, 5, true);      // conv666 -> 384x512
  // ---------------- decoder ----------------
  run_up(bufA, skip4, bufB, 0, 512, 8);       // cat4: 1536x1024
  run_conv(bufB, bufA, N4, 11, 4, true);
  run_conv(bufA, bufB, N4, 12, 4, true);
  run_conv(bufB, bufA, N4, 13, 4, true);      // 1536x256
  run_up(bufA, skip3, bufB, 1, 256, 16);      // cat3: 6144x512
  run_conv(bufB, bufA, N3, 14, 3, true);
  run_conv(bufA, bufB, N3, 15, 3, true);
  run_conv(bufB, bufA, N3, 16, 3, true);      // 6144x128
  run_up(bufA, skip2, bufB, 2, 128, 32);      // cat2: 24576x256
  run_conv(bufB, bufA, N2, 17, 2, true);
  run_conv(bufA, bufB, N2, 18, 2, true);
  run_conv(bufB, bufA, N2, 19, 2, true);      // 24576x64
  run_up(bufA, skip1, bufB, 3, 64, 64);       // cat1: 98304x128
  run_conv(bufB, bufA, N1, 20, 1, true);
  run_conv(bufA, bufB, N1, 21, 1, true);
  run_conv(bufB, out, N1, 22, 1, false);      // conv11: no ReLU
}
